// MultiheadSelfAttention_59261958750787
// MI455X (gfx1250) — matrix operand order repaired and checked by reading, not compiled, not run on hardware
//
#include <hip/hip_runtime.h>
#include <hip/hip_bf16.h>

// ---------------------------------------------------------------------------
// MI455X (gfx1250) multi-head self-attention with RoPE, bf16 WMMA pipeline.
// GEMMs: TDM (tensor_load_to_lds) double-buffered LDS tiles -> ds_load ->
// v_wmma_f32_16x16x32_bf16. Attention: flash-style, WMMA + prefetch.
// ---------------------------------------------------------------------------

typedef __bf16          v16bf __attribute__((ext_vector_type(16)));
typedef float           v8f   __attribute__((ext_vector_type(8)));
typedef unsigned short  v8us  __attribute__((ext_vector_type(8)));
typedef unsigned short  v16us __attribute__((ext_vector_type(16)));
typedef unsigned int    u32x4 __attribute__((ext_vector_type(4)));
typedef int             i32x8 __attribute__((ext_vector_type(8)));
typedef int             i32x4 __attribute__((ext_vector_type(4)));

#define BATCH 2
#define SEQ   2048
#define DM    2048
#define NH    16
#define DK    128
#define MTOT  (BATCH * SEQ)   // 4096 token rows
#define KSTEP 32

// ---- bf16 <-> f32 via bit ops (round-to-nearest-even) ----------------------
__device__ __forceinline__ unsigned short f2us(float f) {
  unsigned u = __float_as_uint(f);
  unsigned r = u + 0x7FFFu + ((u >> 16) & 1u);
  return (unsigned short)(r >> 16);
}
__device__ __forceinline__ float us2f(unsigned short h) {
  return __uint_as_float(((unsigned)h) << 16);
}

// LDS byte offset: flat LDS addresses carry the offset in addr[31:0].
__device__ __forceinline__ unsigned lds_off(const void* p) {
  return (unsigned)(uintptr_t)p;
}

// ---- WMMA fragment loaders (wave32, contiguous k = 16*(lane/16) + s, same in A and B) --------------
// A-matrix 16x32 bf16 from row-major src (ld elements):
//   lane m (0-15):  v16bf[0..15] = K 0..15
//   lane m+16:      v16bf[0..15] = K 16..31
__device__ __forceinline__ v16bf load_a_frag(const unsigned short* src, int ld,
                                             int row0, int k0) {
  int lane = threadIdx.x & 31;
  int m = lane & 15, hi = lane >> 4;
  const unsigned short* p = src + (size_t)(row0 + m) * ld + k0 + hi * 16;
  v8us lo = *(const v8us*)(p);
  v8us hh = *(const v8us*)(p + 8);
  v16us r;
#pragma unroll
  for (int i = 0; i < 8; ++i) { r[i] = lo[i]; r[8 + i] = hh[i]; }
  return __builtin_bit_cast(v16bf, r);
}

// B-matrix 32x16 bf16, given srcT where srcT[n*ld + k] = B[k][n]:
//   lane n (0-15): K 0..15 ; lane n+16: K 16..31
__device__ __forceinline__ v16bf load_bt_frag(const unsigned short* srcT, int ld,
                                              int n0, int k0) {
  int lane = threadIdx.x & 31;
  int n = lane & 15, hi = lane >> 4;
  const unsigned short* p = srcT + (size_t)(n0 + n) * ld + k0 + hi * 16;
  v8us lo = *(const v8us*)(p);
  v8us hh = *(const v8us*)(p + 8);
  v16us r;
#pragma unroll
  for (int i = 0; i < 8; ++i) { r[i] = lo[i]; r[8 + i] = hh[i]; }
  return __builtin_bit_cast(v16bf, r);
}

__device__ __forceinline__ v8f wmma_bf16(v16bf a, v16bf b, v8f c) {
  return __builtin_amdgcn_wmma_f32_16x16x32_bf16(
      /*neg_a=*/false, a, /*neg_b=*/false, b,
      /*c_mod=*/(short)0, c, /*reuse_a=*/false, /*reuse_b=*/false);
}

// ---- Tensor Data Mover: 2D bf16 tile (tile_k x tile_rows) -> LDS -----------
// D# bitfields per CDNA5 ISA 08_async_tensor §8.3/§8.4.
// 6-arg builtin form (clang-23 / therock-10.0 headers):
//   (uint32x4 g0, int32x8 g1, int32x4 g2, int32x4 g3, int32x8 extra, i32 cpol)
__device__ __forceinline__ void tdm_load_2d(unsigned lds_byte_off,
                                            const unsigned short* gsrc,
                                            unsigned tile_k,       // tile_dim0 (contiguous, elems)
                                            unsigned tile_rows,    // tile_dim1
                                            unsigned row_stride,   // tensor_dim0_stride (elems)
                                            unsigned tensor_d0,
                                            unsigned tensor_d1) {
  unsigned long long ga = (unsigned long long)(uintptr_t)gsrc;
  u32x4 g0;
  g0[0] = 1u;                                              // count=1 (valid), user mode
  g0[1] = lds_byte_off;                                    // lds_addr [63:32]
  g0[2] = (unsigned)(ga & 0xFFFFFFFFu);                    // global_addr [95:64]
  g0[3] = (unsigned)((ga >> 32) & 0x1FFFFFFu) | (2u << 30);// addr[56:32], type=2
  i32x8 g1;
  g1[0] = (int)(1u << 16);                                 // data_size=1 (2 bytes)
  g1[1] = (int)((tensor_d0 & 0xFFFFu) << 16);              // tensor_dim0[15:0] @ bit48
  g1[2] = (int)((tensor_d0 >> 16) | ((tensor_d1 & 0xFFFFu) << 16));
  g1[3] = (int)(((tensor_d1 >> 16) & 0xFFFFu) | (tile_k << 16)); // tile_dim0 @ bit112
  g1[4] = (int)(tile_rows & 0xFFFFu);                      // tile_dim1; tile_dim2=0
  g1[5] = (int)row_stride;                                 // tensor_dim0_stride[31:0]
  g1[6] = 0;                                               // stride hi / dim1_stride lo
  g1[7] = 0;
  i32x4 z4 = {0, 0, 0, 0};
  i32x8 z8 = {0, 0, 0, 0, 0, 0, 0, 0};
  __builtin_amdgcn_tensor_load_to_lds(g0, g1, z4, z4, z8, 0);
}

// ---------------------------------------------------------------------------
// fp32 -> bf16 conversion
// ---------------------------------------------------------------------------
__global__ void mha_cvt_bf16_kernel(const float* __restrict__ in,
                                    unsigned short* __restrict__ out, int n) {
  int i = blockIdx.x * blockDim.x + threadIdx.x;
  if (i < n) out[i] = f2us(in[i]);
}

// ---------------------------------------------------------------------------
// GEMM: C[m][e] = sum_d A[m][d] * W[e][d]  (A: MTOT x DM, W: DM x DM, bf16)
// Workgroup = 4 waves -> 64x64 output tile. Per K-step (32): TDM loads a
// 64x32 A tile + 64x32 B tile into LDS (double buffered), waves compute
// 16(M)x64(N) each from LDS fragments.
// mode 0: bf16 [b,h,s,dk] (Q,K) | mode 1: bf16 [b,h,dk,s] (V^T) | mode 2: f32
// ---------------------------------------------------------------------------
__global__ void mha_gemm_kernel(const unsigned short* __restrict__ A,
                                const unsigned short* __restrict__ W,
                                void* __restrict__ out, int mode) {
  __shared__ unsigned short ldsA[2][64 * KSTEP];
  __shared__ unsigned short ldsB[2][64 * KSTEP];

  int wv = threadIdx.x >> 5;
  int lane = threadIdx.x & 31;
  int n = lane & 15, hi = lane >> 4;
  const int tiles_n = DM / 64;                 // 32
  int tile_m = (blockIdx.x / tiles_n) * 64;
  int tile_n = (blockIdx.x % tiles_n) * 64;

  const unsigned short* Abase = A + (size_t)tile_m * DM;
  const unsigned short* Wbase = W + (size_t)tile_n * DM;

  if (wv == 0) {  // prologue: fill buffer 0
    tdm_load_2d(lds_off(&ldsA[0][0]), Abase, KSTEP, 64, DM, DM, MTOT);
    tdm_load_2d(lds_off(&ldsB[0][0]), Wbase, KSTEP, 64, DM, DM, DM);
  }

  v8f acc[4] = {};
  const int NSTEP = DM / KSTEP;                // 64
  for (int i = 0; i < NSTEP; ++i) {
    int cur = i & 1;
    if (wv == 0) {
      if (i + 1 < NSTEP) {
        int k0 = (i + 1) * KSTEP;
        tdm_load_2d(lds_off(&ldsA[cur ^ 1][0]), Abase + k0, KSTEP, 64, DM, DM, MTOT);
        tdm_load_2d(lds_off(&ldsB[cur ^ 1][0]), Wbase + k0, KSTEP, 64, DM, DM, DM);
        __builtin_amdgcn_s_wait_tensorcnt(2);  // current buffer's 2 loads done
      } else {
        __builtin_amdgcn_s_wait_tensorcnt(0);
      }
    }
    __syncthreads();                           // buffer `cur` visible to all
    v16bf a = load_a_frag(&ldsA[cur][0], KSTEP, 16 * wv, 0);
#pragma unroll
    for (int j = 0; j < 4; ++j) {
      v16bf b = load_bt_frag(&ldsB[cur][0], KSTEP, 16 * j, 0);
      acc[j] = wmma_bf16(a, b, acc[j]);
    }
    __syncthreads();                           // done reading before overwrite
  }

  int m0 = tile_m + 16 * wv;
#pragma unroll
  for (int j = 0; j < 4; ++j) {
    int e = tile_n + 16 * j + n;               // output feature (D-frag column)
#pragma unroll
    for (int r = 0; r < 8; ++r) {
      int rg = m0 + r + 8 * hi;                // global token row (D-frag row)
      float v = acc[j][r];
      if (mode == 2) {
        ((float*)out)[(size_t)rg * DM + e] = v;
      } else {
        int b_ = rg >> 11, s = rg & (SEQ - 1);
        int h = e >> 7, d = e & (DK - 1);
        unsigned short* o = (unsigned short*)out;
        if (mode == 0)
          o[((size_t)(b_ * NH + h) * SEQ + s) * DK + d] = f2us(v);
        else
          o[((size_t)(b_ * NH + h) * DK + d) * SEQ + s] = f2us(v);
      }
    }
  }
}

// ---------------------------------------------------------------------------
// RoPE applied in place on bf16 tensor [b,h,s,dk], pair granularity.
// ---------------------------------------------------------------------------
__global__ void mha_rope_kernel(unsigned short* __restrict__ t,
                                const int* __restrict__ pos) {
  size_t idx = (size_t)blockIdx.x * blockDim.x + threadIdx.x;  // pair index
  int i = (int)(idx & 63);
  size_t rest = idx >> 6;                    // (b*NH+h)*SEQ + s
  int s = (int)(rest & (SEQ - 1));
  int b = (int)((rest >> 11) >> 4);
  float p = (float)pos[(size_t)b * SEQ + s];
  float inv_freq = __expf(-__logf(10000.0f) * (float)(2 * i) / (float)DK);
  float rad = p * inv_freq;
  float c = __cosf(rad), sn = __sinf(rad);
  size_t base = idx * 2;
  float x1 = us2f(t[base]), x2 = us2f(t[base + 1]);
  t[base]     = f2us(x1 * c - x2 * sn);
  t[base + 1] = f2us(x1 * sn + x2 * c);
}

// ---------------------------------------------------------------------------
// Flash attention, one wave per (b, h, 16-query tile).
// S^T = K_tile * Q^T so each lane owns one query column; softmax stats need
// only an xor-16 shuffle. Then O^T = V^T * P^T. Next chunk is prefetched.
// ---------------------------------------------------------------------------
__global__ void mha_attn_kernel(const unsigned short* __restrict__ Q,
                                const unsigned short* __restrict__ K,
                                const unsigned short* __restrict__ VT,
                                unsigned short* __restrict__ O) {
  int wid = blockIdx.x * 4 + (threadIdx.x >> 5);  // 4096 waves
  int qtile = wid & 127;
  int h = (wid >> 7) & 15;
  int b = wid >> 11;
  int qbase = qtile * 16;
  int lane = threadIdx.x & 31;
  int n = lane & 15, hi = lane >> 4;

  const unsigned short* Qh = Q + (size_t)(b * NH + h) * SEQ * DK;  // [s][dk]
  const unsigned short* Kh = K + (size_t)(b * NH + h) * SEQ * DK;  // [s][dk]
  const unsigned short* Vh = VT + (size_t)(b * NH + h) * DK * SEQ; // [dk][s]

  v16bf qf[4];
#pragma unroll
  for (int kk = 0; kk < 4; ++kk) qf[kk] = load_bt_frag(Qh, DK, qbase, 32 * kk);

  v8f acc[8] = {};
  float mrun = -__builtin_inff();
  float lrun = 0.0f;
  int qi = qbase + n;
  const float sc = 0.08838834764831845f;  // 1/sqrt(128)

  for (int c = 0; c < qbase + 16; c += 32) {
    if (c + 32 < SEQ) {  // prefetch next key/value chunk (global_prefetch_b8)
      __builtin_prefetch((const void*)(Kh + (size_t)(c + 32) * DK + lane * 4), 0, 1);
      __builtin_prefetch((const void*)(Vh + (size_t)(lane & 7) * SEQ + c + 32), 0, 1);
    }
    v8f s0 = {}, s1 = {};
#pragma unroll
    for (int kk = 0; kk < 4; ++kk) {
      v16bf a0 = load_a_frag(Kh, DK, c, 32 * kk);
      v16bf a1 = load_a_frag(Kh, DK, c + 16, 32 * kk);
      s0 = wmma_bf16(a0, qf[kk], s0);
      s1 = wmma_bf16(a1, qf[kk], s1);
    }
    float t0[8], t1[8];
    float mloc = -__builtin_inff();
#pragma unroll
    for (int r = 0; r < 8; ++r) {
      int k0i = c + r + 8 * hi;
      int k1i = k0i + 16;
      t0[r] = (k0i <= qi) ? s0[r] * sc : -__builtin_inff();
      t1[r] = (k1i <= qi) ? s1[r] * sc : -__builtin_inff();
      mloc = fmaxf(mloc, fmaxf(t0[r], t1[r]));
    }
    mloc = fmaxf(mloc, __shfl_xor(mloc, 16, 32));
    float mnew = fmaxf(mrun, mloc);
    float alpha = __expf(mrun - mnew);
    float rs = 0.0f;
#pragma unroll
    for (int r = 0; r < 8; ++r) {
      t0[r] = __expf(t0[r] - mnew);
      t1[r] = __expf(t1[r] - mnew);
      rs += t0[r] + t1[r];
    }
    rs += __shfl_xor(rs, 16, 32);
    lrun = lrun * alpha + rs;
    mrun = mnew;
#pragma unroll
    for (int t = 0; t < 8; ++t)
#pragma unroll
      for (int r = 0; r < 8; ++r) acc[t][r] *= alpha;

    // P^T B-fragment via xor-16 exchange.
    v16us pu;
#pragma unroll
    for (int r = 0; r < 8; ++r) {
      float send = hi ? t0[r] : t1[r];
      float recv = __shfl_xor(send, 16, 32);
      float lo = hi ? recv : t0[r];
      float hh = hi ? t1[r] : recv;
      pu[r] = f2us(lo);
      pu[8 + r] = f2us(hh);
    }
    v16bf pf = __builtin_bit_cast(v16bf, pu);

#pragma unroll
    for (int t = 0; t < 8; ++t) {
      v16bf vf = load_a_frag(Vh, SEQ, 16 * t, c);
      acc[t] = wmma_bf16(vf, pf, acc[t]);
    }
  }

  float invl = 1.0f / lrun;
  unsigned short* orow = O + ((size_t)b * SEQ + qbase + n) * DM + h * DK;
#pragma unroll
  for (int t = 0; t < 8; ++t) {
    v8us ov;
#pragma unroll
    for (int r = 0; r < 8; ++r) ov[r] = f2us(acc[t][r] * invl);
    *(v8us*)(orow + 16 * t + 8 * hi) = ov;
  }
}

// ---------------------------------------------------------------------------
// Host orchestration. Workspace layout (bytes):
//   0      x_bf16     16MB   [4096][2048]
//   16MB   Wq_bf16     8MB      40MB  Wo_bf16  8MB
//   24MB   Wk_bf16     8MB      48MB  Q_bf16  16MB  [b,h,s,dk]
//   32MB   Wv_bf16     8MB      64MB  K_bf16  16MB  [b,h,s,dk]
//   80MB   VT_bf16    16MB   [b,h,dk,s]
//   96MB   attn_bf16  16MB   [4096][2048]
// ---------------------------------------------------------------------------
extern "C" void kernel_launch(void* const* d_in, const int* in_sizes, int n_in,
                              void* d_out, int out_size, void* d_ws, size_t ws_size,
                              hipStream_t stream) {
  const float* x  = (const float*)d_in[0];
  const int*  pos = (const int*)d_in[1];
  const float* WQ = (const float*)d_in[2];
  const float* WK = (const float*)d_in[3];
  const float* WV = (const float*)d_in[4];
  const float* WO = (const float*)d_in[5];
  float* out = (float*)d_out;

  char* ws = (char*)d_ws;
  const size_t MB = 1024ull * 1024ull;
  unsigned short* xb  = (unsigned short*)(ws + 0 * MB);
  unsigned short* wqb = (unsigned short*)(ws + 16 * MB);
  unsigned short* wkb = (unsigned short*)(ws + 24 * MB);
  unsigned short* wvb = (unsigned short*)(ws + 32 * MB);
  unsigned short* wob = (unsigned short*)(ws + 40 * MB);
  unsigned short* Qb  = (unsigned short*)(ws + 48 * MB);
  unsigned short* Kb  = (unsigned short*)(ws + 64 * MB);
  unsigned short* VTb = (unsigned short*)(ws + 80 * MB);
  unsigned short* ATb = (unsigned short*)(ws + 96 * MB);

  const int nX = MTOT * DM;   // 8,388,608
  const int nW = DM * DM;     // 4,194,304
  mha_cvt_bf16_kernel<<<(nX + 255) / 256, 256, 0, stream>>>(x,  xb,  nX);
  mha_cvt_bf16_kernel<<<(nW + 255) / 256, 256, 0, stream>>>(WQ, wqb, nW);
  mha_cvt_bf16_kernel<<<(nW + 255) / 256, 256, 0, stream>>>(WK, wkb, nW);
  mha_cvt_bf16_kernel<<<(nW + 255) / 256, 256, 0, stream>>>(WV, wvb, nW);
  mha_cvt_bf16_kernel<<<(nW + 255) / 256, 256, 0, stream>>>(WO, wob, nW);

  // Projections: (4096/64)*(2048/64) = 2048 blocks, 4 waves each
  mha_gemm_kernel<<<2048, 128, 0, stream>>>(xb, wqb, (void*)Qb, 0);
  mha_gemm_kernel<<<2048, 128, 0, stream>>>(xb, wkb, (void*)Kb, 0);
  mha_gemm_kernel<<<2048, 128, 0, stream>>>(xb, wvb, (void*)VTb, 1);

  // RoPE on Q and K: BATCH*NH*SEQ*64 = 4,194,304 pairs
  mha_rope_kernel<<<16384, 256, 0, stream>>>(Qb, pos);
  mha_rope_kernel<<<16384, 256, 0, stream>>>(Kb, pos);

  // Flash attention: 4096 waves = 1024 blocks * 4 waves
  mha_attn_kernel<<<1024, 128, 0, stream>>>(Qb, Kb, VTb, ATb);

  // Output projection -> fp32 d_out
  mha_gemm_kernel<<<2048, 128, 0, stream>>>(ATb, wob, (void*)out, 2);
}
